// SparseMambaAttax_52209622450602
// MI455X (gfx1250) — matrix operand order repaired and checked by reading, not compiled, not run on hardware
//
#include <hip/hip_runtime.h>
#include <hip/hip_bf16.h>
#include <math.h>

#define L_SEQ   2048
#define DMODEL  1024
#define NH_ATT  8
#define VHD     64
#define RD      32
#define TOPK    64
#define IDXD    64
#define QLR     128
#define KVLR    128
#define DSTATE  64
#define HDIM    32
#define DINNER  1024
#define NHM     32
#define CONVK   4
#define CONVDIM (DINNER + 2*DSTATE)            /* 1152 */
#define DINPROJ (2*DINNER + 2*DSTATE + NHM)    /* 2208 */
#define NCHUNK  32
#define CHUNKL  64

typedef __attribute__((ext_vector_type(16))) _Float16 v16h;
typedef __attribute__((ext_vector_type(8)))  _Float16 v8h;
typedef __attribute__((ext_vector_type(8)))  float    v8f;

__device__ inline v16h cat16(v8h lo, v8h hi) {
  return __builtin_shufflevector(lo, hi, 0, 1, 2, 3, 4, 5, 6, 7,
                                 8, 9, 10, 11, 12, 13, 14, 15);
}

__device__ inline float silu_f(float v) { return v / (1.0f + expf(-v)); }

// ---------------------------------------------------------------------------
// WMMA GEMM: C[M,N](f32) = A[M,K](f16, row-major) * B, with B given as
// BT[N,K](f16, row-major). K % 32 == 0, M % 128 == 0. One wave -> 16 x NT*16
// output strip starting at column nbase + blockIdx.x*NT*16. NT is a template
// parameter so the inner loop has zero control flow (accumulators stay pinned,
// EXEC stays all-ones around every WMMA).
// ---------------------------------------------------------------------------
template <int NT>
__global__ void __launch_bounds__(256)
k_gemm_f16(const _Float16* __restrict__ A, const _Float16* __restrict__ BT,
           float* __restrict__ C, int M, int N, int K, int nbase) {
  const int lane = threadIdx.x & 31;
  const int wave = threadIdx.x >> 5;
  const int m0 = blockIdx.y * 128 + wave * 16;
  const int n0 = nbase + blockIdx.x * (NT * 16);

  const int mo = lane & 15;
  const int hi = lane >> 4;

  v8f acc[NT];
#pragma unroll
  for (int j = 0; j < NT; ++j) acc[j] = (v8f){};

  const _Float16* arow = A + (size_t)(m0 + mo) * K;
  const _Float16* brow[NT];
#pragma unroll
  for (int j = 0; j < NT; ++j)
    brow[j] = BT + (size_t)(n0 + j * 16 + mo) * K + hi * 16;

#pragma unroll 2
  for (int k0 = 0; k0 < K; k0 += 32) {
    __builtin_prefetch(arow + k0 + 64, 0, 1);     // speculative; OOB dropped
    // A fragment (contiguous k order, same as B): a[i]=A[mo][hi*16+i]
    v16h a = cat16(*(const v8h*)(arow + k0 + hi * 16),
                   *(const v8h*)(arow + k0 + hi * 16 + 8));
#pragma unroll
    for (int j = 0; j < NT; ++j) {
      // B fragment (32x16 layout): b[k'] = B[k0+hi*16+k'][n0+j*16+mo]
      v16h b = cat16(*(const v8h*)(brow[j] + k0),
                     *(const v8h*)(brow[j] + k0 + 8));
      acc[j] = __builtin_amdgcn_wmma_f32_16x16x32_f16(
          false, a, false, b, (short)0, acc[j], false, false);
    }
  }
  // C/D layout: VGPR v -> row m0 + hi*8 + v, col n0 + j*16 + mo
#pragma unroll
  for (int j = 0; j < NT; ++j) {
    float* crow = C + (size_t)(m0 + hi * 8) * N + (n0 + j * 16 + mo);
#pragma unroll
    for (int v = 0; v < 8; ++v) crow[(size_t)v * N] = acc[j][v];
  }
}

// --------------------------- elementwise helpers ---------------------------
__global__ void k_cvt_f16(const float* __restrict__ in, _Float16* __restrict__ out, int n) {
  int i = blockIdx.x * 256 + threadIdx.x;
  if (i < n) out[i] = (_Float16)in[i];
}

// W[K,N] f32 -> WT[N,K] f16
__global__ void k_tr_f16(const float* __restrict__ in, _Float16* __restrict__ out,
                         int K, int N) {
  int k = blockIdx.x * 32 + threadIdx.x;
  int n = blockIdx.y * 8 + threadIdx.y;
  if (k < K && n < N) out[(size_t)n * K + k] = (_Float16)in[(size_t)k * N + n];
}

// depthwise causal conv (K=4) over zxbcdt[:,1024:2176] + bias + SiLU
__global__ void k_conv_silu(const float* __restrict__ zx, const float* __restrict__ cw,
                            const float* __restrict__ cb, float* __restrict__ out) {
  int idx = blockIdx.x * 256 + threadIdx.x;
  if (idx >= L_SEQ * CONVDIM) return;
  int t = idx / CONVDIM, c = idx % CONVDIM;
  float acc = cb[c];
#pragma unroll
  for (int k = 0; k < CONVK; ++k) {
    int tt = t - (CONVK - 1) + k;
    if (tt >= 0) acc += zx[(size_t)tt * DINPROJ + DINNER + c] * cw[c * CONVK + k];
  }
  out[idx] = silu_f(acc);
}

// dt = softplus(raw + bias); da = exp(-dt * exp(A_log))
__global__ void k_dtprep(const float* __restrict__ zx, const float* __restrict__ dt_bias,
                         const float* __restrict__ A_log, float* __restrict__ dt_soft,
                         float* __restrict__ da) {
  int idx = blockIdx.x * 256 + threadIdx.x;
  if (idx >= L_SEQ * NHM) return;
  int t = idx >> 5, h = idx & 31;
  float r = zx[(size_t)t * DINPROJ + 2 * DINNER + 2 * DSTATE + h] + dt_bias[h];
  float dt = (r > 20.0f) ? r : log1pf(expf(r));
  dt_soft[idx] = dt;
  da[idx] = expf(-dt * expf(A_log[h]));
}

// ------------------------- chunked SSM scan (3 pass) -----------------------
__global__ void __launch_bounds__(1024)
k_scan_local(const float* __restrict__ xbc, const float* __restrict__ dt_soft,
             const float* __restrict__ da, float* __restrict__ F) {
  __shared__ float sB[DSTATE], sDt[NHM], sDa[NHM];
  const int c = blockIdx.x, tid = threadIdx.x, h = tid >> 5;
  float st[DSTATE];
#pragma unroll
  for (int s = 0; s < DSTATE; ++s) st[s] = 0.0f;
  for (int t = c * CHUNKL; t < (c + 1) * CHUNKL; ++t) {
    __syncthreads();
    if (tid < 64)       sB[tid] = xbc[(size_t)t * CONVDIM + DINNER + tid];
    else if (tid < 96)  sDt[tid - 64] = dt_soft[t * NHM + (tid - 64)];
    else if (tid < 128) sDa[tid - 96] = da[t * NHM + (tid - 96)];
    __syncthreads();
    float xdt = xbc[(size_t)t * CONVDIM + tid] * sDt[h];
    float dec = sDa[h];
#pragma unroll
    for (int s = 0; s < DSTATE; ++s) st[s] = dec * st[s] + xdt * sB[s];
  }
#pragma unroll
  for (int s = 0; s < DSTATE; ++s)
    F[((size_t)c * 1024 + tid) * DSTATE + s] = st[s];
}

__global__ void k_chunk_prod(const float* __restrict__ da, float* __restrict__ Pc) {
  int idx = blockIdx.x * 256 + threadIdx.x;
  if (idx >= NCHUNK * NHM) return;
  int c = idx / NHM, h = idx % NHM;
  float p = 1.0f;
  for (int t = c * CHUNKL; t < (c + 1) * CHUNKL; ++t) p *= da[t * NHM + h];
  Pc[idx] = p;
}

__global__ void k_scan_combine(const float* __restrict__ F, const float* __restrict__ Pc,
                               float* __restrict__ Sinit) {
  int e = blockIdx.x * 256 + threadIdx.x;          // e = tid*64 + s
  if (e >= 1024 * DSTATE) return;
  int h = (e >> 6) >> 5;
  float s0 = 0.0f;
  for (int c = 0; c < NCHUNK; ++c) {
    Sinit[(size_t)c * 1024 * DSTATE + e] = s0;
    s0 = Pc[c * NHM + h] * s0 + F[(size_t)c * 1024 * DSTATE + e];
  }
}

__global__ void __launch_bounds__(1024)
k_scan_y(const float* __restrict__ xbc, const float* __restrict__ dt_soft,
         const float* __restrict__ da, const float* __restrict__ Sinit,
         const float* __restrict__ Dssm, float* __restrict__ y) {
  __shared__ float sB[DSTATE], sC[DSTATE], sDt[NHM], sDa[NHM];
  const int c = blockIdx.x, tid = threadIdx.x, h = tid >> 5;
  float st[DSTATE];
#pragma unroll
  for (int s = 0; s < DSTATE; ++s)
    st[s] = Sinit[((size_t)c * 1024 + tid) * DSTATE + s];
  const float Dh = Dssm[h];
  for (int t = c * CHUNKL; t < (c + 1) * CHUNKL; ++t) {
    __syncthreads();
    if (tid < 64)       sB[tid] = xbc[(size_t)t * CONVDIM + DINNER + tid];
    else if (tid < 128) sC[tid - 64] = xbc[(size_t)t * CONVDIM + DINNER + DSTATE + (tid - 64)];
    else if (tid < 160) sDt[tid - 128] = dt_soft[t * NHM + (tid - 128)];
    else if (tid < 192) sDa[tid - 160] = da[t * NHM + (tid - 160)];
    __syncthreads();
    float x = xbc[(size_t)t * CONVDIM + tid];
    float xdt = x * sDt[h];
    float dec = sDa[h];
    float acc = Dh * x;
#pragma unroll
    for (int s = 0; s < DSTATE; ++s) {
      st[s] = dec * st[s] + xdt * sB[s];
      acc += st[s] * sC[s];
    }
    y[(size_t)t * DINNER + tid] = acc;
  }
}

// gated RMSNorm -> f16
__global__ void __launch_bounds__(256)
k_gate_norm(const float* __restrict__ yssm, const float* __restrict__ zx,
            const float* __restrict__ nw, _Float16* __restrict__ out) {
  __shared__ float red[256];
  int t = blockIdx.x, tid = threadIdx.x;
  float g[4]; float ss = 0.0f;
#pragma unroll
  for (int i = 0; i < 4; ++i) {
    int c = tid + i * 256;
    float z = zx[(size_t)t * DINPROJ + c];
    float v = yssm[(size_t)t * DINNER + c] * silu_f(z);
    g[i] = v; ss += v * v;
  }
  red[tid] = ss; __syncthreads();
  for (int off = 128; off > 0; off >>= 1) {
    if (tid < off) red[tid] += red[tid + off];
    __syncthreads();
  }
  float scale = rsqrtf(red[0] / (float)DINNER + 1e-5f);
#pragma unroll
  for (int i = 0; i < 4; ++i) {
    int c = tid + i * 256;
    out[(size_t)t * DINNER + c] = (_Float16)(g[i] * scale * nw[c]);
  }
}

// fused causal index scores + top-64 (tie-break lowest index, used -> NaN)
__global__ void __launch_bounds__(256)
k_topk(const float* __restrict__ q_idx, const float* __restrict__ k_idx,
       int* __restrict__ indices) {
  __shared__ float q[IDXD];
  __shared__ float sc[L_SEQ];
  __shared__ float rv[256];
  __shared__ int   ri[256];
  int t = blockIdx.x, tid = threadIdx.x;
  if (tid < IDXD) q[tid] = q_idx[(size_t)t * IDXD + tid];
  __syncthreads();
  for (int s = tid; s < L_SEQ; s += 256) {
    float v;
    if (s <= t) {
      const float* kr = k_idx + (size_t)s * IDXD;
      float acc = 0.0f;
      for (int i = 0; i < IDXD; ++i) acc += q[i] * kr[i];
      v = acc * 0.125f;                       // IDXD^-0.5
    } else v = -__builtin_inff();
    sc[s] = v;
  }
  __syncthreads();
  for (int it = 0; it < TOPK; ++it) {
    float bv = -__builtin_inff(); int bi = 0x7fffffff;
    for (int s = tid; s < L_SEQ; s += 256) {
      float v = sc[s];
      if (v == v && (v > bv || (v == bv && s < bi))) { bv = v; bi = s; }
    }
    rv[tid] = bv; ri[tid] = bi; __syncthreads();
    for (int off = 128; off > 0; off >>= 1) {
      if (tid < off) {
        float ov = rv[tid + off]; int oi = ri[tid + off];
        if (ov > rv[tid] || (ov == rv[tid] && oi < ri[tid])) { rv[tid] = ov; ri[tid] = oi; }
      }
      __syncthreads();
    }
    if (tid == 0) { indices[t * TOPK + it] = ri[0]; sc[ri[0]] = __builtin_nanf(""); }
    __syncthreads();
  }
}

// build q_f / k_f = [content(64) | rope(32)] per head
__global__ void __launch_bounds__(256)
k_rope(const float* __restrict__ qc, const float* __restrict__ qr,
       const float* __restrict__ kvb, const float* __restrict__ krope,
       float* __restrict__ q_f, float* __restrict__ k_f) {
  __shared__ float s16[16], c16[16];
  int t = blockIdx.x, tid = threadIdx.x;
  if (tid < 16) {
    float ang = (float)t * powf(10000.0f, -(float)tid / 16.0f);
    s16[tid] = sinf(ang); c16[tid] = cosf(ang);
  }
  __syncthreads();
  for (int p = tid; p < NH_ATT * 96; p += 256) {
    int h = p / 96, j = p % 96;
    float qv, kv2;
    if (j < VHD) {
      qv  = qc[(size_t)t * 512 + h * 64 + j];
      kv2 = kvb[(size_t)t * 1024 + h * 64 + j];
    } else {
      int j2 = j - VHD, fi = j2 & 15;
      float s = s16[fi], c = c16[fi];
      float q1 = qr[(size_t)t * 256 + h * 32 + fi];
      float q2 = qr[(size_t)t * 256 + h * 32 + 16 + fi];
      float k1 = krope[(size_t)t * 32 + fi];
      float k2 = krope[(size_t)t * 32 + 16 + fi];
      if (j2 < 16) { qv = q1 * c - q2 * s; kv2 = k1 * c - k2 * s; }
      else         { qv = q1 * s + q2 * c; kv2 = k1 * s + k2 * c; }
    }
    q_f[(size_t)t * 768 + p] = qv;
    k_f[(size_t)t * 768 + p] = kv2;
  }
}

// sparse attention over gathered top-64 keys
__global__ void __launch_bounds__(256)
k_attn(const float* __restrict__ q_f, const float* __restrict__ k_f,
       const float* __restrict__ kvb, const int* __restrict__ indices,
       float* __restrict__ out) {
  __shared__ float q[768];
  __shared__ float lg[NH_ATT * TOPK];
  __shared__ int   idx[TOPK];
  int t = blockIdx.x, tid = threadIdx.x;
  if (tid < TOPK) idx[tid] = indices[t * TOPK + tid];
  for (int p = tid; p < 768; p += 256) q[p] = q_f[(size_t)t * 768 + p];
  __syncthreads();
  const float scale = 0.1020620726f;           // 96^-0.5
  for (int p = tid; p < NH_ATT * TOPK; p += 256) {
    int h = p >> 6, k = p & 63;
    const float* kr = k_f + (size_t)idx[k] * 768 + h * 96;
    const float* qh = q + h * 96;
    float acc = 0.0f;
    for (int d = 0; d < 96; ++d) acc += qh[d] * kr[d];
    lg[p] = acc * scale;
  }
  __syncthreads();
  if (tid < NH_ATT) {
    float mx = -__builtin_inff();
    for (int k = 0; k < TOPK; ++k) mx = fmaxf(mx, lg[tid * 64 + k]);
    float sum = 0.0f;
    for (int k = 0; k < TOPK; ++k) { float e = expf(lg[tid * 64 + k] - mx); lg[tid * 64 + k] = e; sum += e; }
    float inv = 1.0f / sum;
    for (int k = 0; k < TOPK; ++k) lg[tid * 64 + k] *= inv;
  }
  __syncthreads();
  for (int p = tid; p < NH_ATT * VHD; p += 256) {
    int h = p >> 6, d = p & 63;
    float acc = 0.0f;
    for (int k = 0; k < TOPK; ++k)
      acc += lg[h * 64 + k] * kvb[(size_t)idx[k] * 1024 + 512 + h * 64 + d];
    out[(size_t)t * 512 + p] = acc;
  }
}

// ---------------------------------------------------------------------------
extern "C" void kernel_launch(void* const* d_in, const int* in_sizes, int n_in,
                              void* d_out, int out_size, void* d_ws, size_t ws_size,
                              hipStream_t stream) {
  (void)in_sizes; (void)n_in; (void)out_size; (void)ws_size;
  const float* x       = (const float*)d_in[0];
  const float* W_in    = (const float*)d_in[1];
  const float* conv_w  = (const float*)d_in[2];
  const float* conv_b  = (const float*)d_in[3];
  const float* dt_bias = (const float*)d_in[4];
  const float* A_log   = (const float*)d_in[5];
  const float* D_ssm   = (const float*)d_in[6];
  const float* norm_w  = (const float*)d_in[7];
  const float* W_mout  = (const float*)d_in[8];
  const float* Wq_idx  = (const float*)d_in[9];
  const float* Wk_idx  = (const float*)d_in[10];
  const float* Wq_down = (const float*)d_in[11];
  const float* Wq_up   = (const float*)d_in[12];
  const float* Wq_rope = (const float*)d_in[13];
  const float* Wkv_down= (const float*)d_in[14];
  const float* Wkv_up  = (const float*)d_in[15];
  const float* Wk_rope = (const float*)d_in[16];
  const float* W_out   = (const float*)d_in[17];

  char* ws = (char*)d_ws;
  size_t off = 0;
  auto alloc = [&](size_t bytes) -> void* {
    void* p = ws + off;
    off += (bytes + 255) & ~(size_t)255;
    return p;
  };
  const size_t L = L_SEQ;
  _Float16* xf16     = (_Float16*)alloc(L * DMODEL * 2);
  _Float16* WinT     = (_Float16*)alloc((size_t)DINPROJ * DMODEL * 2);
  _Float16* WmoutT   = (_Float16*)alloc((size_t)DMODEL * DINNER * 2);
  _Float16* WqidxT   = (_Float16*)alloc((size_t)IDXD * DMODEL * 2);
  _Float16* WkidxT   = (_Float16*)alloc((size_t)IDXD * DMODEL * 2);
  _Float16* WqdownT  = (_Float16*)alloc((size_t)QLR * DMODEL * 2);
  _Float16* WqupT    = (_Float16*)alloc((size_t)(NH_ATT * VHD) * QLR * 2);
  _Float16* WqropeT  = (_Float16*)alloc((size_t)(NH_ATT * RD) * QLR * 2);
  _Float16* WkvdownT = (_Float16*)alloc((size_t)KVLR * DMODEL * 2);
  _Float16* WkvupT   = (_Float16*)alloc((size_t)(2 * NH_ATT * VHD) * KVLR * 2);
  _Float16* WkropeT  = (_Float16*)alloc((size_t)RD * DMODEL * 2);
  _Float16* WoutT    = (_Float16*)alloc((size_t)DMODEL * (NH_ATT * VHD) * 2);
  float* zx      = (float*)alloc(L * DINPROJ * 4);
  float* xbc     = (float*)alloc(L * CONVDIM * 4);
  float* dt_soft = (float*)alloc(L * NHM * 4);
  float* da      = (float*)alloc(L * NHM * 4);
  float* Fst     = (float*)alloc((size_t)NCHUNK * 1024 * DSTATE * 4);
  float* Pc      = (float*)alloc((size_t)NCHUNK * NHM * 4);
  float* Sinit   = (float*)alloc((size_t)NCHUNK * 1024 * DSTATE * 4);
  float* yssm    = (float*)alloc(L * DINNER * 4);
  _Float16* g16  = (_Float16*)alloc(L * DINNER * 2);
  float* xm      = (float*)alloc(L * DMODEL * 4);
  _Float16* xm16 = (_Float16*)alloc(L * DMODEL * 2);
  float* qidx    = (float*)alloc(L * IDXD * 4);
  float* kidx    = (float*)alloc(L * IDXD * 4);
  int*   idxs    = (int*)alloc(L * TOPK * 4);
  float* cq      = (float*)alloc(L * QLR * 4);
  _Float16* cq16 = (_Float16*)alloc(L * QLR * 2);
  float* qcont   = (float*)alloc(L * NH_ATT * VHD * 4);
  float* qrope   = (float*)alloc(L * NH_ATT * RD * 4);
  float* ckv     = (float*)alloc(L * KVLR * 4);
  _Float16* ckv16= (_Float16*)alloc(L * KVLR * 2);
  float* kvb     = (float*)alloc(L * 2 * NH_ATT * VHD * 4);
  float* krope   = (float*)alloc(L * RD * 4);
  float* qf      = (float*)alloc(L * NH_ATT * 96 * 4);
  float* kf      = (float*)alloc(L * NH_ATT * 96 * 4);
  float* attn    = (float*)alloc(L * NH_ATT * VHD * 4);
  _Float16* attn16 = (_Float16*)alloc(L * NH_ATT * VHD * 2);

  auto cvt = [&](const float* in, _Float16* out, int n) {
    k_cvt_f16<<<(n + 255) / 256, 256, 0, stream>>>(in, out, n);
  };
  auto trw = [&](const float* W, _Float16* WT, int K, int N) {
    k_tr_f16<<<dim3((K + 31) / 32, (N + 7) / 8), dim3(32, 8), 0, stream>>>(W, WT, K, N);
  };
  // Dispatch: full 64-wide blocks with NT=4, 32/16-col remainder with NT=2/1.
  auto gemm = [&](const _Float16* A, const _Float16* BT, float* C, int M, int N, int K) {
    int n64 = N / 64;
    if (n64 > 0)
      k_gemm_f16<4><<<dim3(n64, M / 128), 256, 0, stream>>>(A, BT, C, M, N, K, 0);
    int rem = N - n64 * 64;
    if (rem >= 32) {
      k_gemm_f16<2><<<dim3(1, M / 128), 256, 0, stream>>>(A, BT, C, M, N, K, n64 * 64);
      rem -= 32;
    }
    if (rem >= 16)
      k_gemm_f16<1><<<dim3(1, M / 128), 256, 0, stream>>>(A, BT, C, M, N, K, N - 16);
  };

  // --- precision conversion / weight transposes ---
  cvt(x, xf16, L * DMODEL);
  trw(W_in, WinT, DMODEL, DINPROJ);
  trw(W_mout, WmoutT, DINNER, DMODEL);
  trw(Wq_idx, WqidxT, DMODEL, IDXD);
  trw(Wk_idx, WkidxT, DMODEL, IDXD);
  trw(Wq_down, WqdownT, DMODEL, QLR);
  trw(Wq_up, WqupT, QLR, NH_ATT * VHD);
  trw(Wq_rope, WqropeT, QLR, NH_ATT * RD);
  trw(Wkv_down, WkvdownT, DMODEL, KVLR);
  trw(Wkv_up, WkvupT, KVLR, 2 * NH_ATT * VHD);
  trw(Wk_rope, WkropeT, DMODEL, RD);
  trw(W_out, WoutT, NH_ATT * VHD, DMODEL);

  // --- Mamba2 branch ---
  gemm(xf16, WinT, zx, L_SEQ, DINPROJ, DMODEL);
  k_conv_silu<<<(L_SEQ * CONVDIM + 255) / 256, 256, 0, stream>>>(zx, conv_w, conv_b, xbc);
  k_dtprep<<<(L_SEQ * NHM + 255) / 256, 256, 0, stream>>>(zx, dt_bias, A_log, dt_soft, da);
  k_scan_local<<<NCHUNK, 1024, 0, stream>>>(xbc, dt_soft, da, Fst);
  k_chunk_prod<<<(NCHUNK * NHM + 255) / 256, 256, 0, stream>>>(da, Pc);
  k_scan_combine<<<(1024 * DSTATE + 255) / 256, 256, 0, stream>>>(Fst, Pc, Sinit);
  k_scan_y<<<NCHUNK, 1024, 0, stream>>>(xbc, dt_soft, da, Sinit, D_ssm, yssm);
  k_gate_norm<<<L_SEQ, 256, 0, stream>>>(yssm, zx, norm_w, g16);
  gemm(g16, WmoutT, xm, L_SEQ, DMODEL, DINNER);
  cvt(xm, xm16, L * DMODEL);

  // --- indexer + top-k ---
  gemm(xm16, WqidxT, qidx, L_SEQ, IDXD, DMODEL);
  gemm(xf16, WkidxT, kidx, L_SEQ, IDXD, DMODEL);
  k_topk<<<L_SEQ, 256, 0, stream>>>(qidx, kidx, idxs);

  // --- MLA projections ---
  gemm(xf16, WqdownT, cq, L_SEQ, QLR, DMODEL);
  cvt(cq, cq16, L * QLR);
  gemm(cq16, WqupT, qcont, L_SEQ, NH_ATT * VHD, QLR);
  gemm(cq16, WqropeT, qrope, L_SEQ, NH_ATT * RD, QLR);
  gemm(xf16, WkvdownT, ckv, L_SEQ, KVLR, DMODEL);
  cvt(ckv, ckv16, L * KVLR);
  gemm(ckv16, WkvupT, kvb, L_SEQ, 2 * NH_ATT * VHD, KVLR);
  gemm(xf16, WkropeT, krope, L_SEQ, RD, DMODEL);

  // --- RoPE concat, sparse attention, output projection ---
  k_rope<<<L_SEQ, 256, 0, stream>>>(qcont, qrope, kvb, krope, qf, kf);
  k_attn<<<L_SEQ, 256, 0, stream>>>(qf, kf, kvb, idxs, attn);
  cvt(attn, attn16, L * NH_ATT * VHD);
  gemm(attn16, WoutT, (float*)d_out, L_SEQ, DMODEL, NH_ATT * VHD);
}
